// LatticeMP_Block_6356551598646
// MI455X (gfx1250) — matrix operand order repaired and checked by reading, not compiled, not run on hardware
//
#include <hip/hip_runtime.h>
#include <hip/hip_bf16.h>

// ---------------------------------------------------------------------------
// MI455X (gfx1250) implementation.
//
// Dominant cost: A@v = 10000x10000x256 GEMM (51.2 GFLOP) + 400MB of A (f32).
// f32 WMMA (16x16x4) would make this compute-bound ~8x over the HBM time, so
// we convert A to bf16 once (one 400MB read + 200MB write, ~26us @ 23.3TB/s)
// and run everything through v_wmma_f32_16x16x32_bf16 (wave32 WMMA).
// v^T (5MB bf16) stays L2-resident (192MB L2), so A streams exactly once.
//
// K=10000 is not a multiple of the WMMA K-step (32): big-K operands are
// stored with a zero-padded leading dimension of 10016 = 313*32.
//
// The big GEMM optionally runs deterministic split-K=2 (two private f32
// partial buffers + combine kernel, no atomics) to double wave-level
// parallelism while still reading A exactly once. Gated on ws_size.
// ---------------------------------------------------------------------------

typedef __attribute__((ext_vector_type(16))) __bf16 v16bf;
typedef __attribute__((ext_vector_type(8)))  __bf16 v8bf;
typedef __attribute__((ext_vector_type(8)))  float  v8f;

typedef unsigned short u16;
typedef unsigned int   u32;

__device__ __forceinline__ u16 f2bf(float f) {
  u32 u = __float_as_uint(f);
  u32 r = u + 0x7FFFu + ((u >> 16) & 1u);   // round-to-nearest-even
  return (u16)(r >> 16);
}

__device__ __forceinline__ v16bf make16(v8bf lo, v8bf hi) {
  union { v16bf whole; v8bf half[2]; } u;
  u.half[0] = lo; u.half[1] = hi;
  return u.whole;
}

// ---------------------------------------------------------------------------
// Generic bf16 WMMA GEMM:  C[M,N] = epilogue( A[M,K] @ Bt[N,K]^T )
//   A  : bf16 row-major [M, lda]   (lda >= K, K-pad columns are zero)
//   Bt : bf16 row-major [N, ldb]   (B transposed; contiguous-K frag loads)
// Epilogue: (+bias[n]) (+pre_add[m,n]) -> relu -> (+post_add[m,n])
// Outputs (any may be null): f32 (x2), bf16 row-major, bf16 transposed.
// If `partial` != null: raw f32 accumulators are written to
//   partial[blockIdx.z * M*N + m*N + n]  (split-K mode; other outputs unused).
//
// Tiling: 256 threads = 8 waves; wave tile 32x64 (2 M-frags x 4 N-frags of
// 16x16x32), waves arranged 2(M) x 4(N) -> block tile 64 x 256.
// K-loop is branch-free: out-of-range M fragments use row pointers clamped
// to M-1 (loads garbage, stores are guarded).
// Requires: Kpad % 32 == 0, M % 16 == 0, N covered exactly by grid.y*256.
// ---------------------------------------------------------------------------
__global__ __launch_bounds__(256) void wmma_gemm_bf16(
    const u16* __restrict__ A, const u16* __restrict__ Bt,
    const float* __restrict__ bias,
    const float* __restrict__ pre_add,   // added before relu
    const float* __restrict__ post_add,  // added after relu
    float* __restrict__ outf, float* __restrict__ outf2,
    u16* __restrict__ outbf, u16* __restrict__ outbft,
    float* __restrict__ partial,
    int M, int N, int Kpad, int lda, int ldb, int ldbt, int Kslice,
    int do_relu)
{
  const int lane = threadIdx.x & 31;
  const int wave = threadIdx.x >> 5;
  const int ln   = lane & 15;   // element index within 16-wide group
  const int lh   = lane >> 4;   // lane half (0/1)
  const int wm   = wave & 1;
  const int wn   = wave >> 1;
  const int tileM = blockIdx.x * 64  + wm * 32;
  const int tileN = blockIdx.y * 256 + wn * 64;

  const bool mval0 = (tileM      < M);   // M % 16 == 0 -> whole frag valid
  const bool mval1 = (tileM + 16 < M);

  // Clamped A row pointers: branch-free K loop (invalid frags load row M-1).
  int r0 = tileM + ln;        if (r0 > M - 1) r0 = M - 1;
  int r1 = tileM + 16 + ln;   if (r1 > M - 1) r1 = M - 1;
  // A-frag per-lane chunks at k0 + 8*lh and k0 + 16 + 8*lh (ISA 7.12.2).
  const u16* arow0 = A + (long long)r0 * lda + 8 * lh;
  const u16* arow1 = A + (long long)r1 * lda + 8 * lh;
  // B-frag per-lane base: col = tileN + j*16 + ln, chunks at k0+8*lh and k0+16+8*lh
  const u16* bcol[4];
#pragma unroll
  for (int j = 0; j < 4; ++j)
    bcol[j] = Bt + (long long)(tileN + j * 16 + ln) * ldb + 8 * lh;

  // Split-K range for this block.
  const int kc   = blockIdx.z;
  const int kbeg = kc * Kslice;
  int kend = kbeg + Kslice; if (kend > Kpad) kend = Kpad;

  v8f acc[2][4] = {};

  for (int k0 = kbeg; k0 < kend; k0 += 32) {
    v16bf af[2];
    { const v8bf* p = (const v8bf*)(arow0 + k0); af[0] = make16(p[0], p[2]); }
    { const v8bf* p = (const v8bf*)(arow1 + k0); af[1] = make16(p[0], p[2]); }
    v16bf bfr[4];
#pragma unroll
    for (int j = 0; j < 4; ++j) {
      const v8bf* p = (const v8bf*)(bcol[j] + k0);
      bfr[j] = make16(p[0], p[2]);
    }
#pragma unroll
    for (int i = 0; i < 2; ++i)
#pragma unroll
      for (int j = 0; j < 4; ++j)
        acc[i][j] = __builtin_amdgcn_wmma_f32_16x16x32_bf16(
            false, af[i], false, bfr[j], (short)0, acc[i][j], false, false);
  }

  // Epilogue. C/D layout: VGPR r, lane -> (m = r + 8*lh, n = ln) within frag.
  if (partial) {
    float* pz = partial + (long long)kc * M * N;
#pragma unroll
    for (int i = 0; i < 2; ++i) {
      if (!(i == 0 ? mval0 : mval1)) continue;
      const int mbase = tileM + i * 16 + 8 * lh;
#pragma unroll
      for (int j = 0; j < 4; ++j) {
        const int n = tileN + j * 16 + ln;
#pragma unroll
        for (int r = 0; r < 8; ++r)
          pz[(long long)(mbase + r) * N + n] = acc[i][j][r];
      }
    }
    return;
  }

#pragma unroll
  for (int i = 0; i < 2; ++i) {
    if (!(i == 0 ? mval0 : mval1)) continue;
    const int mbase = tileM + i * 16 + 8 * lh;
#pragma unroll
    for (int j = 0; j < 4; ++j) {
      const int n = tileN + j * 16 + ln;
      const float bv = bias ? bias[n] : 0.0f;
#pragma unroll
      for (int r = 0; r < 8; ++r) {
        const long long idx = (long long)(mbase + r) * N + n;
        float x = acc[i][j][r] + bv;
        if (pre_add)  x += pre_add[idx];
        if (do_relu)  x = fmaxf(x, 0.0f);
        if (post_add) x += post_add[idx];
        if (outf)   outf[idx]  = x;
        if (outf2)  outf2[idx] = x;
        if (outbf)  outbf[idx] = f2bf(x);
        if (outbft) outbft[(long long)n * ldbt + (mbase + r)] = f2bf(x);
      }
    }
  }
}

// f32 [R,C] -> bf16 [R,Cpad] with zero pad columns; 4 elements per thread.
// Requires C % 4 == 0 and Cpad % 4 == 0.
__global__ void k_f32_to_bf16_pad(const float* __restrict__ in,
                                  u16* __restrict__ out,
                                  int C, int Cpad, long long n4)
{
  long long i = (long long)blockIdx.x * blockDim.x + threadIdx.x;
  if (i >= n4) return;
  const int cq   = Cpad >> 2;              // float4 groups per padded row
  const int col4 = (int)(i % cq) << 2;
  const long long row = i / cq;
  u32 lo = 0, hi = 0;
  if (col4 < C) {                          // C%4==0: group fully in-range
    float4 x = *(const float4*)(in + row * C + col4);
    lo = (u32)f2bf(x.x) | ((u32)f2bf(x.y) << 16);
    hi = (u32)f2bf(x.z) | ((u32)f2bf(x.w) << 16);
  }
  uint2 r; r.x = lo; r.y = hi;
  *(uint2*)(out + row * Cpad + col4) = r;
}

// in: f32 [Kd, Nd] row-major  ->  out: bf16 [Nd, Kd] row-major (small weights)
__global__ void k_transpose_bf16(const float* __restrict__ in, u16* __restrict__ out,
                                 int Kd, int Nd)
{
  long long idx = (long long)blockIdx.x * blockDim.x + threadIdx.x;
  if (idx >= (long long)Kd * Nd) return;
  int k = (int)(idx % Kd);
  int n = (int)(idx / Kd);
  out[idx] = f2bf(in[(long long)k * Nd + n]);
}

// out_bf = bf16(p0 + p1), 4 elements per thread (deterministic split-K join).
__global__ void k_combine2_bf16(const float* __restrict__ p0,
                                const float* __restrict__ p1,
                                u16* __restrict__ out, long long n4)
{
  long long i = (long long)blockIdx.x * blockDim.x + threadIdx.x;
  if (i >= n4) return;
  float4 a = ((const float4*)p0)[i];
  float4 b = ((const float4*)p1)[i];
  u32 lo = (u32)f2bf(a.x + b.x) | ((u32)f2bf(a.y + b.y) << 16);
  u32 hi = (u32)f2bf(a.z + b.z) | ((u32)f2bf(a.w + b.w) << 16);
  uint2 r; r.x = lo; r.y = hi;
  ((uint2*)out)[i] = r;
}

// h_N[dst] += v[src]; one wave per edge, 256 floats via float4 + f32 atomics.
__global__ __launch_bounds__(256) void k_scatter_add(
    const float* __restrict__ v, const int* __restrict__ src,
    const int* __restrict__ dst, float* __restrict__ out, int E)
{
  int gwave = (int)((blockIdx.x * (long long)blockDim.x + threadIdx.x) >> 5);
  int lane  = threadIdx.x & 31;
  if (gwave >= E) return;
  int s = src[gwave], d = dst[gwave];
  const float4* vi = (const float4*)(v + (long long)s * 256);
  float* o = out + (long long)d * 256;
#pragma unroll
  for (int i = 0; i < 2; ++i) {
    float4 x = vi[lane + 32 * i];
    int b = (lane + 32 * i) * 4;
    atomicAdd(o + b + 0, x.x);
    atomicAdd(o + b + 1, x.y);
    atomicAdd(o + b + 2, x.z);
    atomicAdd(o + b + 3, x.w);
  }
}

extern "C" void kernel_launch(void* const* d_in, const int* in_sizes, int n_in,
                              void* d_out, int out_size, void* d_ws, size_t ws_size,
                              hipStream_t stream)
{
  (void)in_sizes; (void)n_in; (void)out_size;
  const int N = 10000, DIM = 256, HID = 512, OUT = 256, E = 320000;
  const int KPAD = 10016;                  // 313 * 32, zero-padded big-K

  const float* v_c = (const float*)d_in[0];
  const float* Amat= (const float*)d_in[1];
  const int*   src = (const int*)d_in[2];
  const int*   dst = (const int*)d_in[3];
  const float* W1  = (const float*)d_in[4];
  const float* b1  = (const float*)d_in[5];
  const float* W2  = (const float*)d_in[6];
  const float* b2  = (const float*)d_in[7];
  const float* Wl1 = (const float*)d_in[8];
  const float* bl1 = (const float*)d_in[9];
  const float* Wl2 = (const float*)d_in[10];
  const float* bl2 = (const float*)d_in[11];
  float* out = (float*)d_out;

  // Workspace carve-up (256B aligned); ~252 MB base + 20 MB split-K partials.
  char* ws = (char*)d_ws;
  size_t off = 0;
  auto alloc = [&](size_t bytes) -> char* {
    char* p = ws + off;
    off += (bytes + 255) & ~(size_t)255;
    return p;
  };
  u16*  A_bf  = (u16*)alloc((size_t)N * KPAD * 2);    // 200.3 MB, padded rows
  u16*  vc_bf = (u16*)alloc((size_t)N * DIM * 2);
  u16*  W1t   = (u16*)alloc((size_t)HID * DIM * 2);   // [512,256]
  u16*  W2t   = (u16*)alloc((size_t)DIM * HID * 2);   // [256,512]
  u16*  Wl1t  = (u16*)alloc((size_t)OUT * DIM * 2);
  u16*  Wl2t  = (u16*)alloc((size_t)OUT * DIM * 2);
  u16*  h1_bf = (u16*)alloc((size_t)N * HID * 2);
  u16*  v_bf  = (u16*)alloc((size_t)N * DIM * 2);
  u16*  vt_bf = (u16*)alloc((size_t)DIM * KPAD * 2);  // v^T, padded rows
  u16*  Av_bf = (u16*)alloc((size_t)N * OUT * 2);
  float* t1   = (float*)alloc((size_t)N * OUT * 4);
  float* vfin = (float*)alloc((size_t)N * OUT * 4);
  size_t base_need = off;
  float* part0 = (float*)alloc((size_t)N * OUT * 4);  // split-K partials
  float* part1 = (float*)alloc((size_t)N * OUT * 4);
  const bool splitk = (ws_size >= off);               // gate on real ws size

  // ---- precision conversion / weight transposes ----
  {
    long long n4 = (long long)N * KPAD / 4;
    k_f32_to_bf16_pad<<<(unsigned)((n4 + 255) / 256), 256, 0, stream>>>(
        Amat, A_bf, N /*C: A is N x N*/, KPAD, n4);
  }
  {
    long long n4 = (long long)N * DIM / 4;
    k_f32_to_bf16_pad<<<(unsigned)((n4 + 255) / 256), 256, 0, stream>>>(
        v_c, vc_bf, DIM, DIM, n4);
  }
  k_transpose_bf16<<<(DIM * HID + 255) / 256, 256, 0, stream>>>(W1,  W1t,  DIM, HID);
  k_transpose_bf16<<<(DIM * HID + 255) / 256, 256, 0, stream>>>(W2,  W2t,  HID, DIM);
  k_transpose_bf16<<<(DIM * OUT + 255) / 256, 256, 0, stream>>>(Wl1, Wl1t, DIM, OUT);
  k_transpose_bf16<<<(DIM * OUT + 255) / 256, 256, 0, stream>>>(Wl2, Wl2t, DIM, OUT);
  // v^T pad columns (k = 10000..10015) must be exact zeros:
  hipMemsetAsync(vt_bf, 0, (size_t)DIM * KPAD * 2, stream);

  const unsigned gx = (unsigned)((N + 63) / 64);   // 157 row tiles

  // 1) h1 = relu(v_c @ W1 + b1)                    [N,512] bf16
  wmma_gemm_bf16<<<dim3(gx, HID / 256, 1), 256, 0, stream>>>(
      vc_bf, W1t, b1, nullptr, nullptr, nullptr, nullptr, h1_bf, nullptr,
      nullptr, N, HID, DIM, DIM, DIM, 0, DIM, 1);
  // 2) v = relu(v_c + h1 @ W2 + b2)                [N,256] bf16 + bf16^T
  wmma_gemm_bf16<<<dim3(gx, DIM / 256, 1), 256, 0, stream>>>(
      h1_bf, W2t, b2, v_c, nullptr, nullptr, nullptr, v_bf, vt_bf,
      nullptr, N, DIM, HID, HID, HID, KPAD /*ldbt*/, HID, 1);
  // 3) t1 = relu(v @ Wl1 + bl1)                    [N,256] f32
  wmma_gemm_bf16<<<dim3(gx, OUT / 256, 1), 256, 0, stream>>>(
      v_bf, Wl1t, bl1, nullptr, nullptr, t1, nullptr, nullptr, nullptr,
      nullptr, N, OUT, DIM, DIM, DIM, 0, DIM, 1);
  // 4) Av = A @ v  (the big one; A streams once, v^T L2-resident)
  if (splitk) {
    const int KS = ((KPAD / 32 + 1) / 2) * 32;       // 157*32 = 5024
    wmma_gemm_bf16<<<dim3(gx, OUT / 256, 2), 256, 0, stream>>>(
        A_bf, vt_bf, nullptr, nullptr, nullptr, nullptr, nullptr, nullptr,
        nullptr, part0, N, OUT, KPAD, KPAD, KPAD, 0, KS, 0);
    long long n4 = (long long)N * OUT / 4;
    k_combine2_bf16<<<(unsigned)((n4 + 255) / 256), 256, 0, stream>>>(
        part0, part1, Av_bf, n4);
  } else {
    wmma_gemm_bf16<<<dim3(gx, OUT / 256, 1), 256, 0, stream>>>(
        A_bf, vt_bf, nullptr, nullptr, nullptr, nullptr, nullptr, Av_bf,
        nullptr, nullptr, N, OUT, KPAD, KPAD, KPAD, 0, KPAD, 0);
  }
  // 5) vfinal = t1 + relu(Av @ Wl2 + bl2) -> d_out and ws copy
  wmma_gemm_bf16<<<dim3(gx, OUT / 256, 1), 256, 0, stream>>>(
      Av_bf, Wl2t, bl2, nullptr, t1, out, vfin, nullptr, nullptr,
      nullptr, N, OUT, DIM, DIM, DIM, 0, DIM, 1);
  // 6) d_out += segment_sum(vfinal[src] -> dst)
  k_scatter_add<<<(E * 32 + 255) / 256, 256, 0, stream>>>(vfin, src, dst, out, E);
  (void)base_need; (void)part1;
}
